// BroadcastHub_31344671326593
// MI455X (gfx1250) — matrix operand order repaired and checked by reading, not compiled, not run on hardware
//
#include <hip/hip_runtime.h>

#define HUBD 4096
#define BATCHN 4096
#define EXN 8
#define H1N 2048
#define H2N 1024

typedef __attribute__((ext_vector_type(16))) __bf16 v16bf;
typedef __attribute__((ext_vector_type(8)))  float  v8f;

struct BHStats {
    float expected;
    float err_mean;
    float af;
    float bf;
    float probs[4];
    int   level;
    int   pad;
};

union BHFrag { uint4 q[2]; v16bf v; };

__device__ __forceinline__ unsigned short bh_f2bf(float f) {
    unsigned int u = __float_as_uint(f);
    u += 0x7FFFu + ((u >> 16) & 1u);   // round-to-nearest-even
    return (unsigned short)(u >> 16);
}

// Async copy 16B global -> LDS (CDNA5, tracked with ASYNCcnt). GVS addressing:
// global = SADDR(u64) + VADDR(i32); LDS byte address from the VDST VGPR.
__device__ __forceinline__ void bh_async_b128(unsigned lds_off, const void* sbase,
                                              unsigned voff) {
    asm volatile("global_load_async_to_lds_b128 %0, %1, %2"
                 :: "v"(lds_off), "v"(voff), "s"(sbase)
                 : "memory");
}
__device__ __forceinline__ unsigned bh_lds_off(const void* p) {
    // addrspacecast(local->flat) keeps the 0-based LDS offset in addr[31:0]
    return (unsigned)(unsigned long long)p;
}

// ---------------------------------------------------------------------------
// Elementwise fp32 -> bf16 (8 elems / thread)
// ---------------------------------------------------------------------------
__global__ void bh_cvt_bf16(const float* __restrict__ src, unsigned short* __restrict__ dst) {
    size_t i = ((size_t)blockIdx.x * 256 + threadIdx.x) * 8;
    float4 a = *(const float4*)(src + i);
    float4 b = *(const float4*)(src + i + 4);
    ushort4 u, w;
    u.x = bh_f2bf(a.x); u.y = bh_f2bf(a.y); u.z = bh_f2bf(a.z); u.w = bh_f2bf(a.w);
    w.x = bh_f2bf(b.x); w.y = bh_f2bf(b.y); w.z = bh_f2bf(b.z); w.w = bh_f2bf(b.w);
    *(ushort4*)(dst + i) = u;
    *(ushort4*)(dst + i + 4) = w;
}

// ---------------------------------------------------------------------------
// Transpose + fp32->bf16 convert:  dst[C][R] = cvt(src[R][C]), batched over z.
// ---------------------------------------------------------------------------
__global__ void bh_transpose_cvt(const float* __restrict__ src,
                                 unsigned short* __restrict__ dst,
                                 int R, int C) {
    __shared__ unsigned short tile[64 * 65];
    size_t base = (size_t)blockIdx.z * (size_t)R * (size_t)C;
    int tR = blockIdx.y * 64;
    int tC = blockIdx.x * 64;
    int t = threadIdx.x;
#pragma unroll
    for (int i = 0; i < 16; i++) {
        int li = t + i * 256;
        int lr = li >> 6, lc = li & 63;
        float f = src[base + (size_t)(tR + lr) * C + tC + lc];
        tile[lr * 65 + lc] = bh_f2bf(f);
    }
    __syncthreads();
#pragma unroll
    for (int i = 0; i < 16; i++) {
        int li = t + i * 256;
        int lc2 = li >> 6, lr2 = li & 63;
        dst[base + (size_t)(tC + lc2) * R + tR + lr2] = tile[lr2 * 65 + lc2];
    }
}

// ---------------------------------------------------------------------------
// Gate fold: v[e,d] = sum_o Wg[e,o] * Ws[e,o,d]
// ---------------------------------------------------------------------------
__global__ void bh_gatefold(const float* __restrict__ Ws,
                            const float* __restrict__ Wg,
                            float* __restrict__ v) {
    int d = blockIdx.x * 256 + threadIdx.x;
    int e = blockIdx.y;
    const float* wcol = Ws + (size_t)e * HUBD * HUBD + d;
    const float* g = Wg + e * HUBD;
    float acc = 0.f;
    for (int o = 0; o < HUBD; o++) acc += g[o] * wcol[(size_t)o * HUBD];
    v[e * HUBD + d] = acc;
}

__global__ void bh_gatebias(const float* __restrict__ Wg, const float* __restrict__ bs,
                            const float* __restrict__ bg, float* __restrict__ cg,
                            BHStats* st) {
    __shared__ float red[256];
    int t = threadIdx.x;
    if (t < 4) st->probs[t] = 0.f;
    for (int e = 0; e < EXN; e++) {
        float p = 0.f;
#pragma unroll
        for (int i = 0; i < 16; i++) {
            int idx = e * HUBD + t + i * 256;
            p += Wg[idx] * bs[idx];
        }
        red[t] = p; __syncthreads();
        for (int s = 128; s > 0; s >>= 1) { if (t < s) red[t] += red[t + s]; __syncthreads(); }
        if (t == 0) cg[e] = red[0] + bg[e];
        __syncthreads();
    }
}

__global__ void bh_errnet1(const float* __restrict__ We1, const float* __restrict__ be1,
                           const float* __restrict__ hub, float* __restrict__ he) {
    __shared__ float red[256];
    int t = threadIdx.x, j = blockIdx.x;
    float p = 0.f;
#pragma unroll
    for (int i = 0; i < 16; i++) {
        int d = t + i * 256;
        p += We1[(size_t)j * HUBD + d] * hub[d];
    }
    red[t] = p; __syncthreads();
    for (int s = 128; s > 0; s >>= 1) { if (t < s) red[t] += red[t + s]; __syncthreads(); }
    if (t == 0) he[j] = tanhf(red[0] + be1[j]);
}

__global__ void bh_errnet2(const float* __restrict__ We2, const float* __restrict__ be2,
                           const float* __restrict__ he, BHStats* st) {
    __shared__ float red[256];
    int t = threadIdx.x;
    float p = 0.f;
#pragma unroll
    for (int i = 0; i < 4; i++) { int d = t + i * 256; p += he[d] * We2[d]; }
    red[t] = p; __syncthreads();
    for (int s = 128; s > 0; s >>= 1) { if (t < s) red[t] += red[t + s]; __syncthreads(); }
    if (t == 0) st->expected = red[0] + be2[0];
}

__global__ void bh_rowstats(const float* __restrict__ x, float* __restrict__ s1,
                            float* __restrict__ s2) {
    __shared__ float r1[256], r2[256];
    int t = threadIdx.x, b = blockIdx.x;
    float a = 0.f, q = 0.f;
#pragma unroll
    for (int i = 0; i < 16; i++) {
        float f = x[(size_t)b * HUBD + t + i * 256];
        a += f; q += f * f;
    }
    r1[t] = a; r2[t] = q; __syncthreads();
    for (int s = 128; s > 0; s >>= 1) {
        if (t < s) { r1[t] += r1[t + s]; r2[t] += r2[t + s]; }
        __syncthreads();
    }
    if (t == 0) { s1[b] = r1[0]; s2[b] = r2[0]; }
}

__global__ void bh_errmean(const float* __restrict__ s1, const float* __restrict__ s2,
                           BHStats* st) {
    __shared__ float red[256];
    int t = threadIdx.x;
    float e = st->expected;
    float acc = 0.f;
    for (int b = t; b < BATCHN; b += 256) {
        float S1 = s1[b], S2 = s2[b];
        float recon = S2 - 2.f * e * S1 + (float)HUBD * e * e;
        float cur = sqrtf(S2);
        float em = fabsf(e) * 64.f + 1e-7f;   // ||expected|| = |e|*sqrt(4096)
        acc += recon + 0.5f * cur / em;
    }
    red[t] = acc; __syncthreads();
    for (int s = 128; s > 0; s >>= 1) { if (t < s) red[t] += red[t + s]; __syncthreads(); }
    if (t == 0) st->err_mean = red[0] / (float)BATCHN;
}

__global__ void bh_gate(const float* __restrict__ x, const float* __restrict__ v,
                        const float* __restrict__ cg, float* __restrict__ gw) {
    __shared__ float red[256];
    __shared__ float gl[EXN];
    int t = threadIdx.x, b = blockIdx.x;
    float xr[16];
#pragma unroll
    for (int i = 0; i < 16; i++) xr[i] = x[(size_t)b * HUBD + t + i * 256];
    float p[EXN];
#pragma unroll
    for (int e = 0; e < EXN; e++) {
        float a = 0.f;
#pragma unroll
        for (int i = 0; i < 16; i++) a += xr[i] * v[e * HUBD + t + i * 256];
        p[e] = a;
    }
    for (int e = 0; e < EXN; e++) {
        red[t] = p[e]; __syncthreads();
        for (int s = 128; s > 0; s >>= 1) { if (t < s) red[t] += red[t + s]; __syncthreads(); }
        if (t == 0) gl[e] = 1.f / (1.f + __expf(-(red[0] + cg[e])));
        __syncthreads();
    }
    if (t == 0) {
        float m = gl[0];
        for (int e = 1; e < EXN; e++) m = fmaxf(m, gl[e]);
        float s = 0.f, pe[EXN];
        for (int e = 0; e < EXN; e++) { pe[e] = __expf(gl[e] - m); s += pe[e]; }
        for (int e = 0; e < EXN; e++) gw[b * EXN + e] = pe[e] / s;
    }
}

// ---------------------------------------------------------------------------
// MoE GEMM with async double-buffered LDS pipeline.
// combined[b,o] = sum_e gw[b,e] * ( x[b,:]·WsT[e,:,o] + bs[e,o] )
// 1024 chunks = 8 experts x 128 K-steps; per-expert epilogue from LDS arrays.
// ---------------------------------------------------------------------------
__global__ __launch_bounds__(256) void bh_moe_gemm(
    const unsigned short* __restrict__ xb, const unsigned short* __restrict__ WsT,
    const float* __restrict__ bs, const float* __restrict__ gw,
    float* __restrict__ combined, unsigned short* __restrict__ combinedb) {
    __shared__ unsigned short Asm[2][128 * 40];   // [m][k] pad 32->40
    __shared__ unsigned short Bsm[2][32 * 136];   // [k][n] pad 128->136
    __shared__ float gwe[128][EXN];
    __shared__ float bse[EXN][128];

    int t = threadIdx.x;
    int lane = t & 31;
    int wave = t >> 5;
    int wm = wave >> 2;   // 0..1
    int wn = wave & 3;    // 0..3
    int rowbase = blockIdx.y * 128;
    int colbase = blockIdx.x * 128;
    int l15 = lane & 15;
    int hh = lane >> 4;

    // stage per-block gate weights & biases for all experts once
    for (int i = t; i < 128 * EXN; i += 256) {
        int r = i >> 3, e = i & 7;
        gwe[r][e] = gw[(rowbase + r) * EXN + e];
    }
    for (int i = t; i < EXN * 128; i += 256) {
        int e = i >> 7, c = i & 127;
        bse[e][c] = bs[(size_t)e * HUBD + colbase + c];
    }

    unsigned aLds = bh_lds_off(&Asm[0][0]);
    unsigned bLds = bh_lds_off(&Bsm[0][0]);
    const unsigned aStride = 128 * 40 * 2;
    const unsigned bStride = 32 * 136 * 2;
    const unsigned short* aBase = xb + (size_t)rowbase * HUBD;

    // per-thread async slots: A 512 slots (m=s>>2,q=s&3), B 512 slots (k=s>>4,seg=s&15)
    int a0m = t >> 2,        a0q = t & 3;
    int a1m = (t + 256) >> 2, a1q = t & 3;
    int b0k = t >> 4,        b0s = t & 15;
    int b1k = (t + 256) >> 4, b1s = t & 15;

    auto prefetch = [&](int c, int buf) {
        int e = c >> 7, kc = c & 127;
        int k0 = kc * 32;
        const unsigned short* bBase = WsT + (size_t)e * HUBD * HUBD + colbase;
        bh_async_b128(aLds + buf * aStride + a0m * 80 + a0q * 16, aBase,
                      (unsigned)((a0m * HUBD + k0 + a0q * 8) * 2));
        bh_async_b128(aLds + buf * aStride + a1m * 80 + a1q * 16, aBase,
                      (unsigned)((a1m * HUBD + k0 + a1q * 8) * 2));
        bh_async_b128(bLds + buf * bStride + b0k * 272 + b0s * 16, bBase,
                      (unsigned)(((k0 + b0k) * HUBD + b0s * 8) * 2));
        bh_async_b128(bLds + buf * bStride + b1k * 272 + b1s * 16, bBase,
                      (unsigned)(((k0 + b1k) * HUBD + b1s * 8) * 2));
    };

    v8f zero = {};
    v8f comb[4][2], acc[4][2];
#pragma unroll
    for (int mi = 0; mi < 4; mi++)
#pragma unroll
        for (int ni = 0; ni < 2; ni++) { comb[mi][ni] = zero; acc[mi][ni] = zero; }

    const int NC = EXN * (HUBD / 32);   // 1024
    prefetch(0, 0);
    for (int c = 0; c < NC; c++) {
        int cur = c & 1;
        if (c + 1 < NC) {
            prefetch(c + 1, cur ^ 1);
            asm volatile("s_wait_asynccnt 4" ::: "memory");   // current tile resident
        } else {
            asm volatile("s_wait_asynccnt 0" ::: "memory");
        }
        __syncthreads();

        BHFrag af[4], bf2[2];
#pragma unroll
        for (int mi = 0; mi < 4; mi++) {
            int m = wm * 64 + mi * 16 + l15;
            af[mi].q[0] = *(const uint4*)&Asm[cur][m * 40 + hh * 8];
            af[mi].q[1] = *(const uint4*)&Asm[cur][m * 40 + 16 + hh * 8];
        }
#pragma unroll
        for (int ni = 0; ni < 2; ni++) {
            int n0 = wn * 32 + ni * 16;
#pragma unroll
            for (int s = 0; s < 16; s++)
                ((unsigned short*)&bf2[ni])[s] = Bsm[cur][((s & 8) * 2 + hh * 8 + (s & 7)) * 136 + n0 + l15];
        }
#pragma unroll
        for (int mi = 0; mi < 4; mi++)
#pragma unroll
            for (int ni = 0; ni < 2; ni++)
                acc[mi][ni] = __builtin_amdgcn_wmma_f32_16x16x32_bf16(
                    false, af[mi].v, false, bf2[ni].v, (short)0, acc[mi][ni], false, false);
        __syncthreads();

        if ((c & 127) == 127) {           // expert boundary: fold into combined
            int e = c >> 7;
#pragma unroll
            for (int mi = 0; mi < 4; mi++)
#pragma unroll
                for (int ni = 0; ni < 2; ni++) {
                    int cL = wn * 32 + ni * 16 + l15;
                    float bsv = bse[e][cL];
#pragma unroll
                    for (int j = 0; j < 8; j++) {
                        int rL = wm * 64 + mi * 16 + j + hh * 8;
                        comb[mi][ni][j] += gwe[rL][e] * (acc[mi][ni][j] + bsv);
                    }
                    acc[mi][ni] = zero;
                }
        }
    }

#pragma unroll
    for (int mi = 0; mi < 4; mi++)
#pragma unroll
        for (int ni = 0; ni < 2; ni++) {
            int cL = wn * 32 + ni * 16 + l15;
#pragma unroll
            for (int j = 0; j < 8; j++) {
                int rL = wm * 64 + mi * 16 + j + hh * 8;
                size_t idx = (size_t)(rowbase + rL) * HUBD + colbase + cL;
                float vv = comb[mi][ni][j];
                combined[idx] = vv;
                combinedb[idx] = bh_f2bf(vv);   // bf16 copy: A operand of level GEMM
            }
        }
}

// ---------------------------------------------------------------------------
// Level GEMM (async double-buffered): hidden = relu(combined @ Wl1.T + bl1)
// ---------------------------------------------------------------------------
__global__ __launch_bounds__(256) void bh_lvl_gemm(
    const unsigned short* __restrict__ Ab, const unsigned short* __restrict__ BT,
    const float* __restrict__ bias, float* __restrict__ out) {
    __shared__ unsigned short Asm[2][128 * 40];
    __shared__ unsigned short Bsm[2][32 * 136];
    __shared__ float bse[128];

    int t = threadIdx.x;
    int lane = t & 31;
    int wave = t >> 5;
    int wm = wave >> 2;
    int wn = wave & 3;
    int rowbase = blockIdx.y * 128;
    int colbase = blockIdx.x * 128;
    int l15 = lane & 15;
    int hh = lane >> 4;

    if (t < 128) bse[t] = bias[colbase + t];

    unsigned aLds = bh_lds_off(&Asm[0][0]);
    unsigned bLds = bh_lds_off(&Bsm[0][0]);
    const unsigned aStride = 128 * 40 * 2;
    const unsigned bStride = 32 * 136 * 2;
    const unsigned short* aBase = Ab + (size_t)rowbase * HUBD;
    const unsigned short* bBase = BT + colbase;

    int a0m = t >> 2,        a0q = t & 3;
    int a1m = (t + 256) >> 2, a1q = t & 3;
    int b0k = t >> 4,        b0s = t & 15;
    int b1k = (t + 256) >> 4, b1s = t & 15;

    auto prefetch = [&](int kc, int buf) {
        int k0 = kc * 32;
        bh_async_b128(aLds + buf * aStride + a0m * 80 + a0q * 16, aBase,
                      (unsigned)((a0m * HUBD + k0 + a0q * 8) * 2));
        bh_async_b128(aLds + buf * aStride + a1m * 80 + a1q * 16, aBase,
                      (unsigned)((a1m * HUBD + k0 + a1q * 8) * 2));
        bh_async_b128(bLds + buf * bStride + b0k * 272 + b0s * 16, bBase,
                      (unsigned)(((k0 + b0k) * H1N + b0s * 8) * 2));
        bh_async_b128(bLds + buf * bStride + b1k * 272 + b1s * 16, bBase,
                      (unsigned)(((k0 + b1k) * H1N + b1s * 8) * 2));
    };

    v8f zero = {};
    v8f acc[4][2];
#pragma unroll
    for (int mi = 0; mi < 4; mi++)
#pragma unroll
        for (int ni = 0; ni < 2; ni++) acc[mi][ni] = zero;

    const int NC = HUBD / 32;   // 128
    prefetch(0, 0);
    for (int c = 0; c < NC; c++) {
        int cur = c & 1;
        if (c + 1 < NC) {
            prefetch(c + 1, cur ^ 1);
            asm volatile("s_wait_asynccnt 4" ::: "memory");
        } else {
            asm volatile("s_wait_asynccnt 0" ::: "memory");
        }
        __syncthreads();

        BHFrag af[4], bf2[2];
#pragma unroll
        for (int mi = 0; mi < 4; mi++) {
            int m = wm * 64 + mi * 16 + l15;
            af[mi].q[0] = *(const uint4*)&Asm[cur][m * 40 + hh * 8];
            af[mi].q[1] = *(const uint4*)&Asm[cur][m * 40 + 16 + hh * 8];
        }
#pragma unroll
        for (int ni = 0; ni < 2; ni++) {
            int n0 = wn * 32 + ni * 16;
#pragma unroll
            for (int s = 0; s < 16; s++)
                ((unsigned short*)&bf2[ni])[s] = Bsm[cur][((s & 8) * 2 + hh * 8 + (s & 7)) * 136 + n0 + l15];
        }
#pragma unroll
        for (int mi = 0; mi < 4; mi++)
#pragma unroll
            for (int ni = 0; ni < 2; ni++)
                acc[mi][ni] = __builtin_amdgcn_wmma_f32_16x16x32_bf16(
                    false, af[mi].v, false, bf2[ni].v, (short)0, acc[mi][ni], false, false);
        __syncthreads();
    }

#pragma unroll
    for (int mi = 0; mi < 4; mi++)
#pragma unroll
        for (int ni = 0; ni < 2; ni++) {
            int cL = wn * 32 + ni * 16 + l15;
            float bsv = bse[cL];
#pragma unroll
            for (int j = 0; j < 8; j++) {
                int rL = wm * 64 + mi * 16 + j + hh * 8;
                float vv = acc[mi][ni][j] + bsv;
                out[(size_t)(rowbase + rL) * H1N + colbase + cL] = fmaxf(vv, 0.f);
            }
        }
}

__global__ void bh_logits(const float* __restrict__ hidden, const float* __restrict__ Wl2,
                          const float* __restrict__ bl2, BHStats* st) {
    __shared__ float4 red[256];
    int t = threadIdx.x, b = blockIdx.x;
    float l0 = 0, l1 = 0, l2 = 0, l3 = 0;
#pragma unroll
    for (int i = 0; i < 8; i++) {
        int h = t + i * 256;
        float hv = hidden[(size_t)b * H1N + h];
        l0 += hv * Wl2[h];
        l1 += hv * Wl2[H1N + h];
        l2 += hv * Wl2[2 * H1N + h];
        l3 += hv * Wl2[3 * H1N + h];
    }
    red[t] = make_float4(l0, l1, l2, l3); __syncthreads();
    for (int s = 128; s > 0; s >>= 1) {
        if (t < s) {
            float4 a = red[t], bq = red[t + s];
            red[t] = make_float4(a.x + bq.x, a.y + bq.y, a.z + bq.z, a.w + bq.w);
        }
        __syncthreads();
    }
    if (t == 0) {
        float L[4] = { red[0].x + bl2[0], red[0].y + bl2[1], red[0].z + bl2[2], red[0].w + bl2[3] };
        float m = fmaxf(fmaxf(L[0], L[1]), fmaxf(L[2], L[3]));
        float pch[4], s = 0.f;
        for (int c = 0; c < 4; c++) { pch[c] = __expf(L[c] - m); s += pch[c]; }
        for (int c = 0; c < 4; c++) atomicAdd(&st->probs[c], pch[c] / s);
    }
}

__global__ void bh_decide(BHStats* st, int* lvl_out) {
    float mx = -1.f; int ml = 0;
    for (int c = 0; c < 4; c++) {
        float p = st->probs[c] / (float)BATCHN;
        if (p > mx) { mx = p; ml = c; }
    }
    float em = st->err_mean;
    int level;
    if (em > 1.5f) level = 2;
    else if (mx > 0.7f) level = ml;
    else if (em < 0.5f) level = 0;
    else level = 3;
    const float at[4] = { 0.9f, 0.7f, 1.f, 1.f };
    const float bt[4] = { 0.1f, 0.3f, 0.f, 0.f };
    st->af = at[level];
    st->bf = bt[level];
    st->level = level;
    *lvl_out = level;
}

__global__ void bh_outmix(const float* __restrict__ combined, const float* __restrict__ hub,
                          const BHStats* __restrict__ st, float* __restrict__ out) {
    size_t idx = (size_t)blockIdx.x * 256 + threadIdx.x;
    if (idx >= (size_t)BATCHN * HUBD) return;
    out[idx] = st->af * combined[idx] + st->bf * hub[idx & (HUBD - 1)];
}

// ---------------------------------------------------------------------------
extern "C" void kernel_launch(void* const* d_in, const int* in_sizes, int n_in,
                              void* d_out, int out_size, void* d_ws, size_t ws_size,
                              hipStream_t stream) {
    const float* x    = (const float*)d_in[0];
    const float* hub  = (const float*)d_in[1];
    const float* Ws   = (const float*)d_in[2];
    const float* bs   = (const float*)d_in[3];
    const float* Wg   = (const float*)d_in[4];
    const float* bg   = (const float*)d_in[5];
    const float* Wl1  = (const float*)d_in[6];
    const float* bl1  = (const float*)d_in[7];
    const float* Wl2  = (const float*)d_in[8];
    const float* bl2  = (const float*)d_in[9];
    const float* We1  = (const float*)d_in[10];
    const float* be1  = (const float*)d_in[11];
    const float* We2  = (const float*)d_in[12];
    const float* be2  = (const float*)d_in[13];

    char* ws = (char*)d_ws;
    size_t off = 0;
    unsigned short* WsT   = (unsigned short*)(ws + off); off += (size_t)EXN * HUBD * HUBD * 2;
    unsigned short* Wl1T  = (unsigned short*)(ws + off); off += (size_t)HUBD * H1N * 2;
    unsigned short* xb    = (unsigned short*)(ws + off); off += (size_t)BATCHN * HUBD * 2;
    unsigned short* combb = (unsigned short*)(ws + off); off += (size_t)BATCHN * HUBD * 2;
    float* combined = (float*)(ws + off); off += (size_t)BATCHN * HUBD * 4;
    float* hidden   = (float*)(ws + off); off += (size_t)BATCHN * H1N * 4;
    float* v        = (float*)(ws + off); off += (size_t)EXN * HUBD * 4;
    float* gw       = (float*)(ws + off); off += (size_t)BATCHN * EXN * 4;
    float* s1       = (float*)(ws + off); off += (size_t)BATCHN * 4;
    float* s2       = (float*)(ws + off); off += (size_t)BATCHN * 4;
    float* he       = (float*)(ws + off); off += (size_t)H2N * 4;
    float* cg       = (float*)(ws + off); off += 256;
    BHStats* st     = (BHStats*)(ws + off); off += 256;

    // 1. weight/activation conversions (bf16; K-major for WMMA B operand)
    bh_transpose_cvt<<<dim3(64, 64, 8), 256, 0, stream>>>(Ws, WsT, HUBD, HUBD);
    bh_transpose_cvt<<<dim3(64, 32, 1), 256, 0, stream>>>(Wl1, Wl1T, H1N, HUBD);
    bh_cvt_bf16<<<(BATCHN * HUBD) / 2048, 256, 0, stream>>>(x, xb);

    // 2. gate fold + biases + error path scalars
    bh_gatefold<<<dim3(16, 8), 256, 0, stream>>>(Ws, Wg, v);
    bh_gatebias<<<1, 256, 0, stream>>>(Wg, bs, bg, cg, st);
    bh_errnet1<<<H2N, 256, 0, stream>>>(We1, be1, hub, he);
    bh_errnet2<<<1, 256, 0, stream>>>(We2, be2, he, st);
    bh_rowstats<<<BATCHN, 256, 0, stream>>>(x, s1, s2);
    bh_errmean<<<1, 256, 0, stream>>>(s1, s2, st);

    // 3. gate weights
    bh_gate<<<BATCHN, 256, 0, stream>>>(x, v, cg, gw);

    // 4. MoE WMMA GEMM (async double-buffered) -> combined (+bf16 copy)
    bh_moe_gemm<<<dim3(HUBD / 128, BATCHN / 128), 256, 0, stream>>>(xb, WsT, bs, gw,
                                                                    combined, combb);

    // 5. level net hidden layer (WMMA, relu, async double-buffered)
    bh_lvl_gemm<<<dim3(H1N / 128, BATCHN / 128), 256, 0, stream>>>(combb, Wl1T, bl1, hidden);

    // 6. logits + prob accumulation, level decision
    bh_logits<<<BATCHN, 256, 0, stream>>>(hidden, Wl2, bl2, st);
    bh_decide<<<1, 1, 0, stream>>>(st, (int*)d_out + (out_size - 1));

    // 7. output mix
    bh_outmix<<<(BATCHN * HUBD) / 256, 256, 0, stream>>>(combined, hub, st, (float*)d_out);
}
